// SelfAttention_61753039782187
// MI455X (gfx1250) — matrix operand order repaired and checked by reading, not compiled, not run on hardware
//
#include <hip/hip_runtime.h>
#include <hip/hip_bf16.h>

// Problem constants (match reference)
#define B_ 4
#define T_ 2048
#define C_ 1024
#define H_ 16
#define S_ 64

typedef __bf16 bf16;
typedef __attribute__((ext_vector_type(4)))  __bf16 v4bf;
typedef __attribute__((ext_vector_type(8)))  __bf16 v8bf;
typedef __attribute__((ext_vector_type(16))) __bf16 v16bf;
typedef __attribute__((ext_vector_type(4)))  float  v4f;
typedef __attribute__((ext_vector_type(8)))  float  v8f;

static __device__ inline v8f vzero8() {
  v8f z;
#pragma unroll
  for (int i = 0; i < 8; ++i) z[i] = 0.0f;
  return z;
}

static __device__ inline v16bf join16(v8bf lo, v8bf hi) {
  return __builtin_shufflevector(lo, hi, 0,1,2,3,4,5,6,7,8,9,10,11,12,13,14,15);
}

// A-operand fragment, 16x32 K-chunk. ISA 7.12.2 (16-bit A 16x32):
// lane holds row (lane&15); elems 0..7 = K(half*8..+7), elems 8..15 = K(16+half*8..+7)
static __device__ inline v16bf frag_a(const bf16* rowk0, int half) {
  v8bf lo = *(const v8bf*)(rowk0 + half * 8);
  v8bf hi = *(const v8bf*)(rowk0 + 16 + half * 8);
  return join16(lo, hi);
}

// B-operand fragment, 32x16 K-chunk. ISA 7.12.4 (16-bit B):
// lane holds column (lane&15); elems 0..7 = K(half*8..+7), elems 8..15 = K(16+half*8..+7)
static __device__ inline v16bf frag_b(const bf16* colk0, int half) {
  v8bf lo = *(const v8bf*)(colk0 + half * 8);
  v8bf hi = *(const v8bf*)(colk0 + 16 + half * 8);
  return join16(lo, hi);
}

static __device__ inline v8f wmma_bf16(v16bf a, v16bf b, v8f c) {
  // D = A*B + C, fp32 accumulate
  return __builtin_amdgcn_wmma_f32_16x16x32_bf16(false, a, false, b, (short)0, c, false, false);
}

// Issue a 16-byte async DMA: global memory -> LDS, tracked by ASYNCcnt.
// Low 32 bits of a generic pointer to __shared__ are the LDS byte address
// (ISA 10.2: LDS aperture address = {shared_base, lds_offset[31:0]}).
static __device__ inline void async_copy_b128(void* lds_dst, const void* gsrc) {
  uint32_t ldsa = (uint32_t)(uintptr_t)lds_dst;
  uint64_t ga   = (uint64_t)(uintptr_t)gsrc;
  asm volatile("global_load_async_to_lds_b128 %0, %1, off"
               :: "v"(ldsa), "v"(ga) : "memory");
}

static __device__ inline void wait_asynccnt_le1() {
  asm volatile("s_wait_asynccnt 0x1" ::: "memory");
}

// ---------------------------------------------------------------- convert
__global__ __launch_bounds__(256) void cvt_f32_bf16(const float* __restrict__ in,
                                                    bf16* __restrict__ out, int n4) {
  int i = blockIdx.x * 256 + threadIdx.x;
  if (i >= n4) return;
  v4f v = *(const v4f*)(in + (size_t)4 * i);
  v4bf o;
#pragma unroll
  for (int j = 0; j < 4; ++j) o[j] = (bf16)v[j];
  *(v4bf*)(out + (size_t)4 * i) = o;
}

// ---------------------------------------------------------------- GEMM  Y = X * W^T
// X: [M,K] bf16 row-major.  W: [N,K] bf16 row-major (so W^T column n == W row n).
// Block = 256 threads = 8 waves, macro tile 128(M) x 64(N); wave w owns rows m0+16w.
// B strips (64 rows x 32 K) flow through a 4-deep LDS ring filled by
// global_load_async_to_lds_b128, issue depth 2, ONE barrier per K-step:
//   wait asynccnt<=1 (oldest tile landed) -> barrier (publish tile i, free ring
//   slot (i+2)&3, last read at step i-2) -> issue tile i+2 -> 4 WMMAs on tile i.
// mode 0: fp32 out [M,N] + bias      (final projection)
// mode 1: bf16 out [B,H,T,S]         (Q, K)
// mode 2: bf16 out [B,H,S,T]         (V transposed for PV B-operand)
__global__ __launch_bounds__(256) void gemm_bf16_k(const bf16* __restrict__ X,
                                                   const bf16* __restrict__ W,
                                                   const float* __restrict__ bias,
                                                   float* __restrict__ Yf,
                                                   bf16* __restrict__ Yb,
                                                   int M, int N, int K, int mode) {
  __shared__ alignas(16) bf16 lb[4][64][40];  // 32 K-elems/row, padded to 80B rows
  const int wave = threadIdx.x >> 5;
  const int lane = threadIdx.x & 31;
  const int l16  = lane & 15;
  const int half = lane >> 4;
  const int m0   = blockIdx.x * 128 + wave * 16;
  const int n0   = blockIdx.y * 64;

  v8f acc[4];
#pragma unroll
  for (int nt = 0; nt < 4; ++nt) acc[nt] = vzero8();

  const bf16* arow = X + (size_t)(m0 + l16) * K;
  const int   srow = threadIdx.x >> 2;        // 0..63
  const int   schk = (threadIdx.x & 3) * 8;   // 0,8,16,24
  const bf16* wsrc = W + (size_t)(n0 + srow) * K + schk;

  // Prologue: two tiles in flight.
  async_copy_b128(&lb[0][srow][schk], wsrc);
  async_copy_b128(&lb[1][srow][schk], wsrc + 32);

#pragma unroll 4
  for (int tile = 0; tile < 32; ++tile) {     // K = 1024 = 32 tiles of 32
    const int k0 = tile * 32;
    wait_asynccnt_le1();                      // oldest DMA (this tile) complete
    __syncthreads();                          // publish tile; ring slot tile+2 free
    if (tile + 2 < 32)
      async_copy_b128(&lb[(tile + 2) & 3][srow][schk], wsrc + k0 + 64);
    v16bf a = frag_a(arow + k0, half);
#pragma unroll
    for (int nt = 0; nt < 4; ++nt) {
      v16bf b = frag_b(&lb[tile & 3][nt * 16 + l16][0], half);  // ds_load_b128 x2
      acc[nt] = wmma_bf16(a, b, acc[nt]);
    }
  }

  // Epilogue. C/D layout: VGPR i -> row (i + 8*half), col l16 within 16x16 tile.
#pragma unroll
  for (int nt = 0; nt < 4; ++nt) {
#pragma unroll
    for (int i = 0; i < 8; ++i) {
      const int m = m0 + i + 8 * half;
      const int n = n0 + nt * 16 + l16;
      const float v = acc[nt][i];
      if (mode == 0) {
        Yf[(size_t)m * N + n] = v + bias[n];
      } else {
        const int b = m / T_, t = m % T_;
        const int h = n >> 6, s = n & 63;
        if (mode == 1)
          Yb[(((size_t)b * H_ + h) * T_ + t) * S_ + s] = (bf16)v;
        else
          Yb[(((size_t)b * H_ + h) * S_ + s) * T_ + t] = (bf16)v;
      }
    }
  }
}

// ---------------------------------------------------------------- flash attention
// Block = 128 threads = 4 waves; block covers one (b,h) 64-row q band, one wave per
// 16-row q tile. Online softmax in fp32 (log2 domain), bf16 WMMA for QK^T and PV.
__global__ __launch_bounds__(128) void attn_k(const bf16* __restrict__ Q,
                                              const bf16* __restrict__ Kc,
                                              const bf16* __restrict__ Vt,
                                              bf16* __restrict__ O) {
  __shared__ alignas(16) bf16 ps[4][16][40];  // per-wave P staging, padded rows
  const int wave = threadIdx.x >> 5;
  const int lane = threadIdx.x & 31;
  const int l16  = lane & 15;
  const int half = lane >> 4;
  const int nq64 = T_ / 64;
  const int q64  = blockIdx.x % nq64;
  const int bh   = blockIdx.x / nq64;
  const int h    = bh % H_;
  const int b    = bh / H_;
  const int qbase = q64 * 64 + wave * 16;

  const bf16* Qb = Q  + (size_t)(b * H_ + h) * T_ * S_;
  const bf16* Kb = Kc + (size_t)(b * H_ + h) * T_ * S_;
  const bf16* Vb = Vt + (size_t)(b * H_ + h) * S_ * T_;

  // Q fragments for this 16-row tile (K-dim = S = 64 -> two 32-chunks), hoisted.
  v16bf q0 = frag_a(Qb + (size_t)(qbase + l16) * S_,      half);
  v16bf q1 = frag_a(Qb + (size_t)(qbase + l16) * S_ + 32, half);

  v8f acc[4];
#pragma unroll
  for (int nt = 0; nt < 4; ++nt) acc[nt] = vzero8();
  float mrow[8], lrow[8];
#pragma unroll
  for (int i = 0; i < 8; ++i) { mrow[i] = -1e30f; lrow[i] = 0.0f; }

  const float sc = 0.03125f * 1.44269504088896f;  // (1/sqrt(C)) * log2(e)
  const int nsteps = (qbase + 16 + 31) >> 5;      // 32 keys per step, causal bound
  bf16 (*pw)[40] = ps[wave];

  for (int st = 0; st < nsteps; ++st) {
    const int jb = st * 32;

    // Prefetch next step's K rows into the cache hierarchy.
    if (jb + 32 + lane < T_)
      __builtin_prefetch(Kb + (size_t)(jb + 32 + lane) * S_, 0, 1);

    // Scores: two 16x16 tiles over keys [jb, jb+32)
    v8f s0 = vzero8(), s1 = vzero8();
    {
      const bf16* kr0 = Kb + (size_t)(jb + l16) * S_;
      const bf16* kr1 = Kb + (size_t)(jb + 16 + l16) * S_;
      s0 = wmma_bf16(q0, frag_b(kr0,      half), s0);
      s0 = wmma_bf16(q1, frag_b(kr0 + 32, half), s0);
      s1 = wmma_bf16(q0, frag_b(kr1,      half), s1);
      s1 = wmma_bf16(q1, frag_b(kr1 + 32, half), s1);
    }

    // Scale, causal mask, online softmax update (per-row stats across 16 lanes).
    float corr[8];
#pragma unroll
    for (int i = 0; i < 8; ++i) {
      const int row = qbase + i + 8 * half;
      float a0 = s0[i] * sc;
      float a1 = s1[i] * sc;
      if (jb + l16      > row) a0 = -1e30f;
      if (jb + 16 + l16 > row) a1 = -1e30f;
      float mx = fmaxf(a0, a1);
#pragma unroll
      for (int o = 1; o < 16; o <<= 1) mx = fmaxf(mx, __shfl_xor(mx, o, 32));
      const float mnew = fmaxf(mrow[i], mx);
      const float c  = __builtin_amdgcn_exp2f(mrow[i] - mnew);
      const float p0 = __builtin_amdgcn_exp2f(a0 - mnew);
      const float p1 = __builtin_amdgcn_exp2f(a1 - mnew);
      float rs = p0 + p1;
#pragma unroll
      for (int o = 1; o < 16; o <<= 1) rs += __shfl_xor(rs, o, 32);
      lrow[i] = lrow[i] * c + rs;
      mrow[i] = mnew;
      corr[i] = c;
      s0[i] = p0;
      s1[i] = p1;
    }
#pragma unroll
    for (int nt = 0; nt < 4; ++nt)
#pragma unroll
      for (int i = 0; i < 8; ++i) acc[nt][i] *= corr[i];

    // Transpose P (C-layout -> A-layout) through this wave's private LDS slice.
#pragma unroll
    for (int i = 0; i < 8; ++i) {
      pw[i + 8 * half][l16]      = (bf16)s0[i];
      pw[i + 8 * half][16 + l16] = (bf16)s1[i];
    }
    asm volatile("s_wait_dscnt 0" ::: "memory");   // same-wave DS store->load fence
    v16bf pf = frag_a(&pw[l16][0], half);

    // acc += P * V : B-operand columns are rows of Vt, contiguous over keys.
#pragma unroll
    for (int nt = 0; nt < 4; ++nt) {
      v16bf vf = frag_b(Vb + (size_t)(nt * 16 + l16) * T_ + jb, half);
      acc[nt] = wmma_bf16(pf, vf, acc[nt]);
    }
  }

  // Normalize and store: O is [B,T,C] bf16 with column h*64+s.
  const size_t obase = ((size_t)b * T_ + qbase) * C_ + (size_t)h * S_;
#pragma unroll
  for (int i = 0; i < 8; ++i) {
    const float inv = 1.0f / lrow[i];
    const int row = i + 8 * half;
#pragma unroll
    for (int nt = 0; nt < 4; ++nt)
      O[obase + (size_t)row * C_ + nt * 16 + l16] = (bf16)(acc[nt][i] * inv);
  }
}

// ---------------------------------------------------------------- launch
extern "C" void kernel_launch(void* const* d_in, const int* in_sizes, int n_in,
                              void* d_out, int out_size, void* d_ws, size_t ws_size,
                              hipStream_t stream) {
  (void)in_sizes; (void)n_in; (void)out_size; (void)ws_size;
  const float* x  = (const float*)d_in[0];
  const float* Wk = (const float*)d_in[1];
  const float* Wq = (const float*)d_in[2];
  const float* Wv = (const float*)d_in[3];
  const float* Wp = (const float*)d_in[4];
  const float* bp = (const float*)d_in[5];
  float* out = (float*)d_out;

  const size_t MK = (size_t)B_ * T_ * C_;  // 8,388,608 elems
  const size_t WW = (size_t)C_ * C_;       // 1,048,576 elems
  bf16* ws  = (bf16*)d_ws;
  bf16* xb  = ws;            // x   bf16
  bf16* wkb = xb  + MK;
  bf16* wqb = wkb + WW;
  bf16* wvb = wqb + WW;
  bf16* wpb = wvb + WW;
  bf16* Qb  = wpb + WW;      // [B,H,T,S]
  bf16* Kb  = Qb  + MK;      // [B,H,T,S]
  bf16* Vtb = Kb  + MK;      // [B,H,S,T]
  bf16* Ab  = Vtb + MK;      // attention out [B,T,C]

  const int M = B_ * T_, N = C_, K = C_;

  cvt_f32_bf16<<<(int)(MK / 4 / 256), 256, 0, stream>>>(x,  xb,  (int)(MK / 4));
  cvt_f32_bf16<<<(int)(WW / 4 / 256), 256, 0, stream>>>(Wk, wkb, (int)(WW / 4));
  cvt_f32_bf16<<<(int)(WW / 4 / 256), 256, 0, stream>>>(Wq, wqb, (int)(WW / 4));
  cvt_f32_bf16<<<(int)(WW / 4 / 256), 256, 0, stream>>>(Wv, wvb, (int)(WW / 4));
  cvt_f32_bf16<<<(int)(WW / 4 / 256), 256, 0, stream>>>(Wp, wpb, (int)(WW / 4));

  dim3 gg(M / 128, N / 64);
  gemm_bf16_k<<<gg, 256, 0, stream>>>(xb, wqb, nullptr, nullptr, Qb,  M, N, K, 1);
  gemm_bf16_k<<<gg, 256, 0, stream>>>(xb, wkb, nullptr, nullptr, Kb,  M, N, K, 1);
  gemm_bf16_k<<<gg, 256, 0, stream>>>(xb, wvb, nullptr, nullptr, Vtb, M, N, K, 2);

  attn_k<<<B_ * H_ * (T_ / 64), 128, 0, stream>>>(Qb, Kb, Vtb, Ab);

  gemm_bf16_k<<<gg, 256, 0, stream>>>(Ab, wpb, bp, out, nullptr, M, N, K, 0);
}
